// AttentiveStatPool_28260884808411
// MI455X (gfx1250) — matrix operand order repaired and checked by reading, not compiled, not run on hardware
//
#include <hip/hip_runtime.h>
#include <hip/hip_bf16.h>

typedef __attribute__((ext_vector_type(16))) __bf16 v16bf;
typedef __attribute__((ext_vector_type(8)))  float  v8f;
typedef unsigned short u16;
typedef unsigned int   u32;
typedef unsigned long long u64;

#define EPS 1e-4f
#define LOG2E 1.4426950408889634f

static constexpr int B_  = 32;
static constexpr int C_  = 1536;
static constexpr int T_  = 2000;
static constexpr int BOT = 128;
static constexpr int C3  = 4608;

union Frag { int4 q[2]; v16bf v; };

__device__ __forceinline__ u16 f2bf(float f) {
  __bf16 h = (__bf16)f;
  return __builtin_bit_cast(u16, h);
}

__device__ __forceinline__ float wsum(float v) {
#pragma unroll
  for (int o = 16; o > 0; o >>= 1) v += __shfl_xor(v, o, 32);
  return v;
}

// Async global->LDS copy (bf16 tiles, no conversion needed): CDNA5 path,
// tracked on ASYNCcnt. Inline asm per bridge-doc guidance (portable vs builtin
// arity differences). LDS operand = byte offset from LDS base (low 32 bits of
// a generic pointer to __shared__).
__device__ __forceinline__ void async_ld_b128(void* lds_ptr, const void* gptr) {
  u32 lo = (u32)(uintptr_t)lds_ptr;
  u64 ga = (u64)(uintptr_t)gptr;
  asm volatile("global_load_async_to_lds_b128 %0, %1, off"
               :: "v"(lo), "v"(ga) : "memory");
}
__device__ __forceinline__ void wait_async0() {
  asm volatile("s_wait_asynccnt 0" ::: "memory");
}

// ---- Kernel 0: pre-convert weights to bf16 (W2 pre-scaled by log2(e)) ----
__global__ __launch_bounds__(256) void k_convert(const float* __restrict__ W1,
                                                 const float* __restrict__ W2,
                                                 u16* __restrict__ w1a,
                                                 u16* __restrict__ w2b) {
  int i = blockIdx.x * 256 + threadIdx.x;
  if (i < BOT * C_) {
    int o = i / C_, c = i - o * C_;
    w1a[i] = f2bf(W1[(size_t)o * C3 + c]);        // W1[:, :C] only
    w2b[i] = f2bf(W2[i] * LOG2E);                 // exp -> exp2 domain
  }
}

// ---- Kernel 1: per-(b,c) mean / unbiased std over T (float4 streaming) ----
__global__ __launch_bounds__(256) void k_stats(const float* __restrict__ x,
                                               float* __restrict__ mean,
                                               float* __restrict__ stdv) {
  __shared__ float ss[8], sq[8];
  size_t row = blockIdx.x;                        // row = b*C + c
  const float4* p4 = (const float4*)(x + row * (size_t)T_);
  float s = 0.f, q = 0.f;
  for (int i = threadIdx.x; i < T_ / 4; i += 256) {
    float4 v = p4[i];
    s += v.x + v.y + v.z + v.w;
    q += v.x * v.x + v.y * v.y + v.z * v.z + v.w * v.w;
  }
  s = wsum(s); q = wsum(q);
  int w = threadIdx.x >> 5, l = threadIdx.x & 31;
  if (l == 0) { ss[w] = s; sq[w] = q; }
  __syncthreads();
  if (threadIdx.x == 0) {
    float S = 0.f, Q = 0.f;
#pragma unroll
    for (int i = 0; i < 8; ++i) { S += ss[i]; Q += sq[i]; }
    float m   = S / (float)T_;
    float var = (Q - (float)T_ * m * m) / (float)(T_ - 1);
    mean[row] = m;
    stdv[row] = sqrtf(fmaxf(var, EPS));
  }
}

// ---- Kernel 2: rank-1 collapse of the mean/std thirds of W1 + b1 ----
__global__ __launch_bounds__(256) void k_cbias(const float* __restrict__ W1,
                                               const float* __restrict__ b1,
                                               const float* __restrict__ mean,
                                               const float* __restrict__ stdv,
                                               float* __restrict__ cbias) {
  __shared__ float ss[8];
  int b = blockIdx.x >> 7, o = blockIdx.x & 127;
  const float* wm = W1 + (size_t)o * C3 + C_;
  const float* ws = W1 + (size_t)o * C3 + 2 * C_;
  const float* mp = mean + (size_t)b * C_;
  const float* sp = stdv + (size_t)b * C_;
  float acc = 0.f;
  for (int c = threadIdx.x; c < C_; c += 256)
    acc += wm[c] * mp[c] + ws[c] * sp[c];
  acc = wsum(acc);
  int w = threadIdx.x >> 5, l = threadIdx.x & 31;
  if (l == 0) ss[w] = acc;
  __syncthreads();
  if (threadIdx.x == 0) {
    float S = 0.f;
#pragma unroll
    for (int i = 0; i < 8; ++i) S += ss[i];
    cbias[blockIdx.x] = S + b1[o];
  }
}

// ---- Kernel 3: h[b][t][o] = relu(W1a @ x + cbias), bf16 WMMA ----
// block = 8 waves; tile M=128(o) x N=16(t); K staged 64-wide, double-buffered.
__global__ __launch_bounds__(256) void k_gemm1(const float* __restrict__ x,
                                               const u16* __restrict__ w1a,
                                               const float* __restrict__ cbias,
                                               u16* __restrict__ ht) {
  __shared__ u16 Xs[2][16 * 72];   // [t][k] bf16, row stride 72 halves (144B)
  int b  = blockIdx.x / (T_ / 16);
  int t0 = (blockIdx.x % (T_ / 16)) * 16;
  int tid = threadIdx.x;
  int w = tid >> 5, l = tid & 31;
  int hlf = l >> 4, ln = l & 15;
  int o0 = w * 16;

  auto stage = [&](int buf, int k0) {   // 64k x 16t tile: 512 k-pairs, 2/thread
#pragma unroll
    for (int i = 0; i < 2; ++i) {
      int p = tid + i * 256;
      int kr2 = p >> 4, tc = p & 15;
      const float* gp = x + ((size_t)b * C_ + (k0 + kr2 * 2)) * T_ + t0 + tc;
      u32 pk = (u32)f2bf(gp[0]) | ((u32)f2bf(gp[T_]) << 16);
      *reinterpret_cast<u32*>(&Xs[buf][tc * 72 + kr2 * 2]) = pk;
    }
  };

  v8f acc = {};
  stage(0, 0);
  int cur = 0;
  for (int kc = 0; kc < C_ / 64; ++kc) {
    __syncthreads();
    if (kc + 1 < C_ / 64) stage(cur ^ 1, (kc + 1) * 64);   // overlap next stage
#pragma unroll
    for (int s = 0; s < 2; ++s) {
      Frag fa, fb;
      size_t ho = (size_t)(o0 + ln) * C_ + kc * 64 + s * 32 + (hlf ? 8 : 0);
      fa.q[0] = *reinterpret_cast<const int4*>(w1a + ho);
      fa.q[1] = *reinterpret_cast<const int4*>(w1a + ho + 16);
      const u16* xb = &Xs[cur][ln * 72 + s * 32 + (hlf ? 8 : 0)];
      fb.q[0] = *reinterpret_cast<const int4*>(xb);
      fb.q[1] = *reinterpret_cast<const int4*>(xb + 16);
      acc = __builtin_amdgcn_wmma_f32_16x16x32_bf16(false, fa.v, false, fb.v,
                                                    (short)0, acc, false, false);
    }
    cur ^= 1;
  }
  // epilogue: + cbias, relu, pack bf16 pairs, store h[b][t][o]
  const float* cb = cbias + b * BOT + o0 + hlf * 8;
  u16* hrow = ht + ((size_t)b * T_ + t0 + ln) * BOT + o0 + hlf * 8;
#pragma unroll
  for (int r = 0; r < 8; r += 2) {
    float v0 = fmaxf(acc[r]     + cb[r],     0.f);
    float v1 = fmaxf(acc[r + 1] + cb[r + 1], 0.f);
    u32 pk = (u32)f2bf(v0) | ((u32)f2bf(v1) << 16);
    *reinterpret_cast<u32*>(hrow + r) = pk;
  }
}

// ---- Kernel 4: fused W2 @ h -> online softmax(base2) over T -> stats ----
// block = 8 waves, owns (b, 128-channel tile); wave w owns 16 channels.
// A(M=t,K=o)=h^T tile, B(K=o,N=c)=W2 tile, D(M=t,N=c).
// W2 tile + double-buffered h tile staged via global_load_async_to_lds_b128.
__global__ __launch_bounds__(256) void k_attn(const float* __restrict__ x,
                                              const u16* __restrict__ w2b,
                                              const u16* __restrict__ ht,
                                              float* __restrict__ out) {
  __shared__ u16 Ws[128 * 136];        // W2 tile [c][o], stride 136 halves
  __shared__ u16 Hs[2][16 * 136];      // h tile  [t][o], double buffered
  int b  = blockIdx.x / (C_ / 128);
  int c0 = (blockIdx.x % (C_ / 128)) * 128;
  int tid = threadIdx.x;
  int w = tid >> 5, l = tid & 31;
  int hlf = l >> 4, ln = l & 15;

  // async-stage W2 tile once (128x128 bf16 = 2048 int4, 8/thread)
#pragma unroll
  for (int i = 0; i < 8; ++i) {
    int g = tid + i * 256;
    int row = g >> 4, c8 = g & 15;
    async_ld_b128(&Ws[row * 136 + c8 * 8],
                  w2b + (size_t)(c0 + row) * BOT + c8 * 8);
  }
  int hrow = tid >> 4, hc8 = tid & 15;           // this thread's h-tile slot
  async_ld_b128(&Hs[0][hrow * 136 + hc8 * 8],    // preload tile 0
                ht + ((size_t)b * T_ + hrow) * BOT + hc8 * 8);

  float m = -1e30f, S = 0.f, Sx = 0.f, Sxx = 0.f;
  int c_lane = c0 + w * 16 + ln;                 // this lane's channel
  const float* xp = x + ((size_t)b * C_ + c_lane) * T_ + hlf * 8;
  int cur = 0;
  for (int tt = 0; tt < T_ / 16; ++tt) {
    int t0 = tt * 16;
    wait_async0();                               // this wave's copies landed
    __syncthreads();                             // everyone's tile visible
    if (tt + 1 < T_ / 16)                        // overlap next h tile
      async_ld_b128(&Hs[cur ^ 1][hrow * 136 + hc8 * 8],
                    ht + ((size_t)b * T_ + t0 + 16 + hrow) * BOT + hc8 * 8);
    v8f lg = {};
#pragma unroll
    for (int kc = 0; kc < 4; ++kc) {             // K = BOT = 4*32
      Frag fa, fb;
      const u16* ap = &Hs[cur][ln * 136 + kc * 32 + (hlf ? 8 : 0)];
      fa.q[0] = *reinterpret_cast<const int4*>(ap);
      fa.q[1] = *reinterpret_cast<const int4*>(ap + 16);
      const u16* bp = &Ws[(w * 16 + ln) * 136 + kc * 32 + (hlf ? 8 : 0)];
      fb.q[0] = *reinterpret_cast<const int4*>(bp);
      fb.q[1] = *reinterpret_cast<const int4*>(bp + 16);
      lg = __builtin_amdgcn_wmma_f32_16x16x32_bf16(false, fa.v, false, fb.v,
                                                   (short)0, lg, false, false);
    }
    // x for t = t0 + hlf*8 + r  (matches D-matrix row layout)
    float4 x0 = *reinterpret_cast<const float4*>(xp + t0);
    float4 x1 = *reinterpret_cast<const float4*>(xp + t0 + 4);
    __builtin_prefetch(xp + t0 + 16, 0, 1);
    float xa[8] = {x0.x, x0.y, x0.z, x0.w, x1.x, x1.y, x1.z, x1.w};
    float tm = lg[0];
#pragma unroll
    for (int r = 1; r < 8; ++r) tm = fmaxf(tm, lg[r]);
    float nm = fmaxf(m, tm);
    float sc = exp2f(m - nm);                    // logits already in log2 domain
    S *= sc; Sx *= sc; Sxx *= sc; m = nm;
#pragma unroll
    for (int r = 0; r < 8; ++r) {
      float e  = exp2f(lg[r] - m);
      float xv = xa[r];
      S += e; Sx += e * xv; Sxx += e * xv * xv;
    }
    cur ^= 1;
  }
  // combine lane pair (hlf 0/1) handling the same channel
  float mo   = __shfl_xor(m,   16, 32);
  float So   = __shfl_xor(S,   16, 32);
  float Sxo  = __shfl_xor(Sx,  16, 32);
  float Sxxo = __shfl_xor(Sxx, 16, 32);
  float nm = fmaxf(m, mo);
  float a0 = exp2f(m - nm), a1 = exp2f(mo - nm);
  float St   = S   * a0 + So   * a1;
  float Sxt  = Sx  * a0 + Sxo  * a1;
  float Sxxt = Sxx * a0 + Sxxo * a1;
  if (hlf == 0) {
    float wm  = Sxt / St;
    float wsd = sqrtf(fmaxf(Sxxt / St - wm * wm, EPS));
    out[(size_t)b * (2 * C_) + c_lane]      = wm;
    out[(size_t)b * (2 * C_) + C_ + c_lane] = wsd;
  }
}

extern "C" void kernel_launch(void* const* d_in, const int* in_sizes, int n_in,
                              void* d_out, int out_size, void* d_ws, size_t ws_size,
                              hipStream_t stream) {
  const float* x  = (const float*)d_in[0];
  const float* W1 = (const float*)d_in[1];
  const float* b1 = (const float*)d_in[2];
  const float* W2 = (const float*)d_in[3];
  // d_in[4] = b2 is a no-op under softmax (shift invariance) -> dropped.
  char* ws = (char*)d_ws;
  float* mean  = (float*)(ws);                 // B*C f32      (196608 B)
  float* stdv  = (float*)(ws + 196608);        // B*C f32
  float* cbias = (float*)(ws + 393216);        // B*BOT f32    (16384 B)
  u16*   w1a   = (u16*)(ws + 409600);          // BOT*C bf16   (393216 B)
  u16*   w2b   = (u16*)(ws + 802816);          // C*BOT bf16   (393216 B)
  u16*   ht    = (u16*)(ws + 1196032);         // B*T*BOT bf16 (16384000 B)
  float* out = (float*)d_out;

  k_convert<<<(BOT * C_ + 255) / 256, 256, 0, stream>>>(W1, W2, w1a, w2b);
  k_stats  <<<B_ * C_,          256, 0, stream>>>(x, mean, stdv);
  k_cbias  <<<B_ * BOT,         256, 0, stream>>>(W1, b1, mean, stdv, cbias);
  k_gemm1  <<<B_ * (T_ / 16),   256, 0, stream>>>(x, w1a, cbias, ht);
  k_attn   <<<B_ * (C_ / 128),  256, 0, stream>>>(x, w2b, ht, out);
}
